// ManifoldGuidedAttention_71940702208344
// MI455X (gfx1250) — matrix operand order repaired and checked by reading, not compiled, not run on hardware
//
#include <hip/hip_runtime.h>
#include <hip/hip_bf16.h>

typedef __attribute__((ext_vector_type(16))) _Float16 v16h;
typedef __attribute__((ext_vector_type(8)))  _Float16 v8h;
typedef __attribute__((ext_vector_type(2)))  _Float16 v2h;
typedef __attribute__((ext_vector_type(8)))  float    v8f;
typedef __attribute__((ext_vector_type(4)))  unsigned int v4u;
typedef __attribute__((ext_vector_type(8)))  int      v8i;
typedef __attribute__((ext_vector_type(4)))  int      v4i;

#define B_   4
#define S_   2048
#define D_   1024
#define NS_  64
#define H_   16
#define HD_  64
#define SCALE_ 0.125f          // HD^-0.5
#define LN_EPS_ 1e-5f
#define KQ_  (D_ + 2 * NS_)    // 1152
#define KK_  (D_ + NS_)        // 1088

#define TDM_AVAILABLE __has_builtin(__builtin_amdgcn_tensor_load_to_lds)

// ---------------------------------------------------------------------------
// Fragment loaders (CDNA5 wave32 WMMA 16x16x32 f16 layouts, ISA 7.12.2)
// ---------------------------------------------------------------------------
__device__ inline v8f v8f_zero() {
    v8f r;
#pragma unroll
    for (int i = 0; i < 8; ++i) r[i] = 0.0f;
    return r;
}

// A matrix 16x32 (MxK): lane%16 = M row; K per VGPR i: (i>>2)*16+(lane>>4)*8+(i&3)*2
__device__ inline v16h load_a_frag(const _Float16* __restrict__ base, int ld, int lane) {
    v16h a;
    const int m  = lane & 15;
    const int kh = lane >> 4;
    const _Float16* rp = base + (size_t)m * ld;
#pragma unroll
    for (int i = 0; i < 8; ++i) {
        const int kb = ((i >> 2) << 4) + kh * 8 + ((i & 3) << 1);
        const v2h p = *reinterpret_cast<const v2h*>(rp + kb);
        a[2 * i]     = p.x;
        a[2 * i + 1] = p.y;
    }
    return a;
}

// B matrix 32x16 (KxN) with memory rows = N dimension (B[k][n] = base[n*ld+k]):
// lane%16 = N col; K per VGPR i: (i>>2)*16+(lane>>4)*8+(i&3)*2 (same as A)
__device__ inline v16h load_b_frag(const _Float16* __restrict__ base, int ld, int lane) {
    v16h b;
    const int n  = lane & 15;
    const int kh = lane >> 4;
    const _Float16* rp = base + (size_t)n * ld;
#pragma unroll
    for (int i = 0; i < 8; ++i) {
        const v2h p = *reinterpret_cast<const v2h*>(rp + ((i >> 2) << 4) + kh * 8 + ((i & 3) << 1));
        b[2 * i]     = p.x;
        b[2 * i + 1] = p.y;
    }
    return b;
}

// ---------------------------------------------------------------------------
// Tensor Data Mover: async 64x64 f16 tile Global -> LDS (ISA ch. 8, D# layout)
// ---------------------------------------------------------------------------
#if TDM_AVAILABLE
__device__ inline void tdm_load_tile64x64_f16(unsigned lds_byte_off,
                                              const _Float16* gsrc,
                                              unsigned row_stride_elems) {
    const unsigned long long ga = (unsigned long long)(size_t)gsrc;
    v4u g0;
    g0[0] = 1u;                                           // count=1 (valid), user mode
    g0[1] = lds_byte_off;                                 // lds_addr [63:32]
    g0[2] = (unsigned)ga;                                 // global_addr [95:64]
    g0[3] = (unsigned)((ga >> 32) & 0x01FFFFFFull)        // global_addr [120:96]
            | (2u << 30);                                 // type=2 ("image") [127:126]
    v8i g1;
    g1[0] = 0x00010000;                                   // data_size=1 (2B), no multicast
    g1[1] = (int)(64u << 16);                             // tensor_dim0 = 64
    g1[2] = (int)(64u << 16);                             // tensor_dim1 = 64
    g1[3] = (int)(64u << 16);                             // tile_dim0 = 64
    g1[4] = 64;                                           // tile_dim1 = 64, tile_dim2 = 0
    g1[5] = (int)row_stride_elems;                        // tensor_dim0_stride (2B units)
    g1[6] = 0;
    g1[7] = 0;
    const v4i z4 = {0, 0, 0, 0};
#if defined(__clang_major__) && __clang_major__ >= 23
    const v8i z8 = {0, 0, 0, 0, 0, 0, 0, 0};
    __builtin_amdgcn_tensor_load_to_lds(g0, g1, z4, z4, z8, 0);
#else
    __builtin_amdgcn_tensor_load_to_lds(g0, g1, z4, z4, 0);
#endif
}

__device__ inline void tdm_wait_all() {
#if __has_builtin(__builtin_amdgcn_s_wait_tensorcnt)
    __builtin_amdgcn_s_wait_tensorcnt(0);
#else
    asm volatile("s_wait_tensorcnt 0x0" ::: "memory");
#endif
}
#endif

__device__ inline unsigned lds_offset_of(const void* p) {
    // generic address of LDS: low 32 bits are the LDS byte offset
    return (unsigned)(size_t)p;
}

// ---------------------------------------------------------------------------
// Pack / convert kernels
// ---------------------------------------------------------------------------
__global__ void cvt_f16_kernel(const float* __restrict__ src, _Float16* __restrict__ dst, int n) {
    int i = blockIdx.x * blockDim.x + threadIdx.x;
    if (i < n) dst[i] = (_Float16)src[i];
}

__global__ void pack_xq_kernel(const float* __restrict__ h, const float* __restrict__ scn,
                               const float* __restrict__ gv, _Float16* __restrict__ xq, int n) {
    int i = blockIdx.x * blockDim.x + threadIdx.x;
    if (i >= n) return;
    const int row = i / KQ_, c = i % KQ_;
    float v;
    if (c < D_)            v = h[(size_t)row * D_ + c];
    else if (c < D_ + NS_) v = scn[(size_t)row * NS_ + (c - D_)];
    else                   v = gv[(size_t)row * NS_ + (c - D_ - NS_)];
    xq[i] = (_Float16)v;
}

__global__ void pack_xk_kernel(const float* __restrict__ h, const float* __restrict__ scn,
                               _Float16* __restrict__ xk, int n) {
    int i = blockIdx.x * blockDim.x + threadIdx.x;
    if (i >= n) return;
    const int row = i / KK_, c = i % KK_;
    const float v = (c < D_) ? h[(size_t)row * D_ + c] : scn[(size_t)row * NS_ + (c - D_)];
    xk[i] = (_Float16)v;
}

// ep = normalize(scn + 0.4*gv); emit f16 copies of scn and ep. One wave per row.
__global__ void ep_scn_kernel(const float* __restrict__ scn, const float* __restrict__ gv,
                              _Float16* __restrict__ scn16, _Float16* __restrict__ ep16) {
    const int row = blockIdx.x;           // b*S + s
    const int l = threadIdx.x;            // 0..31
    const float* sr = scn + (size_t)row * NS_;
    const float* gr = gv  + (size_t)row * NS_;
    const float s0 = sr[l], s1 = sr[l + 32];
    const float e0 = s0 + 0.4f * gr[l];
    const float e1 = s1 + 0.4f * gr[l + 32];
    float ss = e0 * e0 + e1 * e1;
#pragma unroll
    for (int m = 1; m < 32; m <<= 1) ss += __shfl_xor(ss, m, 32);
    const float inv = 1.0f / fmaxf(sqrtf(ss), 1e-12f);
    scn16[(size_t)row * NS_ + l]      = (_Float16)s0;
    scn16[(size_t)row * NS_ + l + 32] = (_Float16)s1;
    ep16[(size_t)row * NS_ + l]       = (_Float16)(e0 * inv);
    ep16[(size_t)row * NS_ + l + 32]  = (_Float16)(e1 * inv);
}

// ---------------------------------------------------------------------------
// Generic WMMA GEMM: acc[m,n] = sum_k X[m,k]*W[n,k]
// OMODE 0: OutT store row-major [M,N] (+optional f32 addend)
// OMODE 1: f16 store transposed per-head: vt[b][head][d][s]  (for V^T)
// Block 128 (4 waves); wave -> 16 rows x 64 cols; grid (N/64, M/64)
// ---------------------------------------------------------------------------
template <typename OutT, int OMODE>
__global__ __launch_bounds__(128) void wmma_gemm_xwT(const _Float16* __restrict__ X,
                                                     const _Float16* __restrict__ W,
                                                     OutT* __restrict__ out,
                                                     const float* __restrict__ addend,
                                                     int M, int N, int Kdim) {
    const int lane = threadIdx.x & 31, w = threadIdx.x >> 5;
    const int nlane = lane & 15, kh = lane >> 4;
    const int m0 = blockIdx.y * 64 + w * 16;
    const int n0 = blockIdx.x * 64;
    v8f acc[4];
#pragma unroll
    for (int nt = 0; nt < 4; ++nt) acc[nt] = v8f_zero();

    for (int k0 = 0; k0 < Kdim; k0 += 32) {
        const v16h a = load_a_frag(X + (size_t)m0 * Kdim + k0, Kdim, lane);
#pragma unroll
        for (int nt = 0; nt < 4; ++nt) {
            const v16h b = load_b_frag(W + (size_t)(n0 + nt * 16) * Kdim + k0, Kdim, lane);
            acc[nt] = __builtin_amdgcn_wmma_f32_16x16x32_f16(false, a, false, b,
                                                             (short)0, acc[nt], false, false);
        }
    }
#pragma unroll
    for (int nt = 0; nt < 4; ++nt)
#pragma unroll
        for (int j = 0; j < 8; ++j) {
            const int row = m0 + j + 8 * kh;
            const int col = n0 + nt * 16 + nlane;
            float v = acc[nt][j];
            if (OMODE == 0) {
                if (addend) v += addend[(size_t)row * N + col];
                out[(size_t)row * N + col] = (OutT)v;
            } else {
                // V^T layout [B][H][HD][S]
                const int bb = row / S_, s = row % S_;
                const int head = col / HD_, d = col % HD_;
                out[(((size_t)bb * H_ + head) * HD_ + d) * S_ + s] = (OutT)v;
            }
        }
}

// ---------------------------------------------------------------------------
// Bias: prox = scn.scn^T; epsim = ep.scn^T; bias = prox*(mw + nw*epsim)
// ---------------------------------------------------------------------------
__global__ __launch_bounds__(128) void bias_kernel(const _Float16* __restrict__ scn16,
                                                   const _Float16* __restrict__ ep16,
                                                   const float* __restrict__ mgp,
                                                   const float* __restrict__ navp,
                                                   float* __restrict__ biasOut) {
    const int lane = threadIdx.x & 31, w = threadIdx.x >> 5;
    const int nlane = lane & 15, kh = lane >> 4;
    const int b  = blockIdx.z;
    const int t0 = blockIdx.x * 64;
    const int s0 = blockIdx.y * 64 + w * 16;

    v8f ap[4], ae[4];
#pragma unroll
    for (int nt = 0; nt < 4; ++nt) { ap[nt] = v8f_zero(); ae[nt] = v8f_zero(); }

#pragma unroll
    for (int k0 = 0; k0 < NS_; k0 += 32) {
        const v16h as = load_a_frag(scn16 + ((size_t)b * S_ + s0) * NS_ + k0, NS_, lane);
        const v16h aE = load_a_frag(ep16  + ((size_t)b * S_ + s0) * NS_ + k0, NS_, lane);
#pragma unroll
        for (int nt = 0; nt < 4; ++nt) {
            const v16h bf = load_b_frag(scn16 + ((size_t)b * S_ + t0 + nt * 16) * NS_ + k0,
                                        NS_, lane);
            ap[nt] = __builtin_amdgcn_wmma_f32_16x16x32_f16(false, as, false, bf,
                                                            (short)0, ap[nt], false, false);
            ae[nt] = __builtin_amdgcn_wmma_f32_16x16x32_f16(false, aE, false, bf,
                                                            (short)0, ae[nt], false, false);
        }
    }
    const float mw = 0.05f + 0.95f / (1.0f + __expf(-mgp[0]));
    const float nw = 0.05f + 0.95f / (1.0f + __expf(-navp[0]));
#pragma unroll
    for (int nt = 0; nt < 4; ++nt)
#pragma unroll
        for (int j = 0; j < 8; ++j) {
            const int srow = s0 + j + 8 * kh;
            const int tcol = t0 + nt * 16 + nlane;
            biasOut[((size_t)b * S_ + srow) * S_ + tcol] = ap[nt][j] * (mw + nw * ae[nt][j]);
        }
}

// ---------------------------------------------------------------------------
// Flash attention. grid (S/64, H, B); block 128 (4 waves, 16 q-rows each).
// K and V^T tiles staged in LDS by the Tensor Data Mover (double-buffered,
// TENSORcnt waits); P round-trips through per-wave LDS for the A-layout.
// ---------------------------------------------------------------------------
__global__ __launch_bounds__(128) void flash_attn_kernel(const _Float16* __restrict__ q16,
                                                         const _Float16* __restrict__ k16,
                                                         const _Float16* __restrict__ vt16,
                                                         const float* __restrict__ biasBuf,
                                                         const unsigned char* __restrict__ maskBuf,
                                                         _Float16* __restrict__ o16) {
    __shared__ _Float16 kbuf[2][64 * 64];   // K tile: rows t, cols d
    __shared__ _Float16 vbuf[2][64 * 64];   // V^T tile: rows d, cols t
    __shared__ _Float16 p_lds[4 * 16 * 64]; // per-wave private 16x64 P tile

    const int lane = threadIdx.x & 31, w = threadIdx.x >> 5;
    const int nlane = lane & 15, kh = lane >> 4;
    const int b = blockIdx.z, head = blockIdx.y;
    const int s0 = blockIdx.x * 64 + w * 16;
    const int NT = S_ / 64;

    const _Float16* kbase  = k16  + ((size_t)b * S_) * D_ + head * HD_;       // row pitch D_
    const _Float16* vtbase = vt16 + (((size_t)b * H_ + head) * HD_) * S_;     // row pitch S_

    // Q fragments (16 rows x 64 k), loaded once
    v16h qf[2];
#pragma unroll
    for (int kk = 0; kk < 2; ++kk)
        qf[kk] = load_a_frag(q16 + ((size_t)b * S_ + s0) * D_ + head * HD_ + kk * 32, D_, lane);

    float mst[8], lst[8];
#pragma unroll
    for (int j = 0; j < 8; ++j) { mst[j] = -3.0e38f; lst[j] = 0.0f; }
    v8f o[4];
#pragma unroll
    for (int nt = 0; nt < 4; ++nt) o[nt] = v8f_zero();

    _Float16* pw = p_lds + w * 16 * 64;

#if TDM_AVAILABLE
    if (w == 0) {
        tdm_load_tile64x64_f16(lds_offset_of(&kbuf[0][0]), kbase, D_);
        tdm_load_tile64x64_f16(lds_offset_of(&vbuf[0][0]), vtbase, S_);
    }
#endif

    for (int it = 0; it < NT; ++it) {
        const int t0 = it * 64;
        const int cur = it & 1;

#if TDM_AVAILABLE
        if (w == 0) tdm_wait_all();     // tile(it) landed in buf[cur]
        __syncthreads();                // visible to all waves; buf[cur^1] reads done
        if (w == 0 && it + 1 < NT) {    // prefetch tile(it+1) into the other buffer
            tdm_load_tile64x64_f16(lds_offset_of(&kbuf[cur ^ 1][0]),
                                   kbase + (size_t)(t0 + 64) * D_, D_);
            tdm_load_tile64x64_f16(lds_offset_of(&vbuf[cur ^ 1][0]),
                                   vtbase + (t0 + 64), S_);
        }
#else
        __syncthreads();                // previous iteration's reads complete
        for (int c = threadIdx.x; c < 512; c += 128) {
            const int r = c >> 3, col8 = (c & 7) * 8;
            *(v8h*)&kbuf[cur][r * 64 + col8] =
                *(const v8h*)(kbase + (size_t)(t0 + r) * D_ + col8);
            *(v8h*)&vbuf[cur][r * 64 + col8] =
                *(const v8h*)(vtbase + (size_t)r * S_ + t0 + col8);
        }
        __syncthreads();
#endif

        // ---- scores = Q @ K^T  (B[k=d][n=t] = Ktile[t][d], contiguous in LDS) ----
        v8f sc[4];
#pragma unroll
        for (int nt = 0; nt < 4; ++nt) sc[nt] = v8f_zero();
#pragma unroll
        for (int kk = 0; kk < 2; ++kk)
#pragma unroll
            for (int nt = 0; nt < 4; ++nt) {
                const v16h bf = load_b_frag(&kbuf[cur][(nt * 16) * 64 + kk * 32], 64, lane);
                sc[nt] = __builtin_amdgcn_wmma_f32_16x16x32_f16(false, qf[kk], false, bf,
                                                                (short)0, sc[nt], false, false);
            }
        // ---- scale + bias + mask + clip ----
#pragma unroll
        for (int nt = 0; nt < 4; ++nt)
#pragma unroll
            for (int j = 0; j < 8; ++j) {
                const int r = s0 + j + 8 * kh;
                const int t = t0 + nt * 16 + nlane;
                float v = sc[nt][j] * SCALE_ + biasBuf[((size_t)b * S_ + r) * S_ + t];
                if (maskBuf[(size_t)r * S_ + t]) v = -10000.0f;
                sc[nt][j] = fminf(fmaxf(v, -50.0f), 50.0f);
            }
        // ---- online softmax ----
        float tm[8];
#pragma unroll
        for (int j = 0; j < 8; ++j)
            tm[j] = fmaxf(fmaxf(sc[0][j], sc[1][j]), fmaxf(sc[2][j], sc[3][j]));
#pragma unroll
        for (int m = 1; m < 16; m <<= 1)
#pragma unroll
            for (int j = 0; j < 8; ++j) tm[j] = fmaxf(tm[j], __shfl_xor(tm[j], m, 32));

        float alpha[8], rs[8];
#pragma unroll
        for (int j = 0; j < 8; ++j) {
            const float mnew = fmaxf(mst[j], tm[j]);
            alpha[j] = __expf(mst[j] - mnew);
            mst[j] = mnew;
            rs[j] = 0.0f;
        }
#pragma unroll
        for (int nt = 0; nt < 4; ++nt)
#pragma unroll
            for (int j = 0; j < 8; ++j) {
                const float p = __expf(sc[nt][j] - mst[j]);
                sc[nt][j] = p;
                rs[j] += p;
            }
#pragma unroll
        for (int m = 1; m < 16; m <<= 1)
#pragma unroll
            for (int j = 0; j < 8; ++j) rs[j] += __shfl_xor(rs[j], m, 32);
#pragma unroll
        for (int j = 0; j < 8; ++j) lst[j] = lst[j] * alpha[j] + rs[j];
#pragma unroll
        for (int nt = 0; nt < 4; ++nt)
#pragma unroll
            for (int j = 0; j < 8; ++j) o[nt][j] *= alpha[j];

        // ---- P (C layout) -> LDS -> A fragments ----
#pragma unroll
        for (int nt = 0; nt < 4; ++nt)
#pragma unroll
            for (int j = 0; j < 8; ++j)
                pw[(j + 8 * kh) * 64 + nt * 16 + nlane] = (_Float16)sc[nt][j];

        // ---- O += P @ V  (B[k=t][n=d] = Vttile[d][t], contiguous in LDS) ----
#pragma unroll
        for (int kk = 0; kk < 2; ++kk) {
            const v16h pa = load_a_frag(pw + kk * 32, 64, lane);
#pragma unroll
            for (int nt = 0; nt < 4; ++nt) {
                const v16h bf = load_b_frag(&vbuf[cur][(nt * 16) * 64 + kk * 32], 64, lane);
                o[nt] = __builtin_amdgcn_wmma_f32_16x16x32_f16(false, pa, false, bf,
                                                               (short)0, o[nt], false, false);
            }
        }
    }
    // ---- normalize + store ----
#pragma unroll
    for (int nt = 0; nt < 4; ++nt)
#pragma unroll
        for (int j = 0; j < 8; ++j) {
            const int row = s0 + j + 8 * kh;
            const int col = head * HD_ + nt * 16 + nlane;
            o16[((size_t)b * S_ + row) * D_ + col] = (_Float16)(o[nt][j] / lst[j]);
        }
}

// ---------------------------------------------------------------------------
// LayerNorm over last dim (1024), one block per row
// ---------------------------------------------------------------------------
__global__ __launch_bounds__(256) void ln_kernel(const float* __restrict__ z,
                                                 const float* __restrict__ gamma,
                                                 const float* __restrict__ beta,
                                                 float* __restrict__ out) {
    __shared__ float s1[256], s2[256];
    const int row = blockIdx.x;
    const float* zr = z + (size_t)row * D_;
    float a = 0.0f, b2 = 0.0f;
    for (int c = threadIdx.x; c < D_; c += 256) {
        const float v = zr[c];
        a += v; b2 += v * v;
    }
    s1[threadIdx.x] = a; s2[threadIdx.x] = b2;
    __syncthreads();
    for (int off = 128; off > 0; off >>= 1) {
        if (threadIdx.x < off) {
            s1[threadIdx.x] += s1[threadIdx.x + off];
            s2[threadIdx.x] += s2[threadIdx.x + off];
        }
        __syncthreads();
    }
    const float mu = s1[0] * (1.0f / D_);
    const float var = s2[0] * (1.0f / D_) - mu * mu;
    const float r = rsqrtf(var + LN_EPS_);
    for (int c = threadIdx.x; c < D_; c += 256)
        out[(size_t)row * D_ + c] = (zr[c] - mu) * r * gamma[c] + beta[c];
}

// ---------------------------------------------------------------------------
// Host launcher
// ---------------------------------------------------------------------------
extern "C" void kernel_launch(void* const* d_in, const int* in_sizes, int n_in,
                              void* d_out, int out_size, void* d_ws, size_t ws_size,
                              hipStream_t stream) {
    (void)in_sizes; (void)n_in; (void)out_size; (void)ws_size;
    const float* h    = (const float*)d_in[0];
    const float* scn  = (const float*)d_in[1];
    const float* gv   = (const float*)d_in[2];
    const unsigned char* mask = (const unsigned char*)d_in[3];
    const float* Wq   = (const float*)d_in[4];
    const float* Wk   = (const float*)d_in[5];
    const float* Wv   = (const float*)d_in[6];
    const float* Wout = (const float*)d_in[7];
    const float* ln_g = (const float*)d_in[8];
    const float* ln_b = (const float*)d_in[9];
    const float* mg   = (const float*)d_in[10];
    const float* nav  = (const float*)d_in[11];
    float* out = (float*)d_out;

    char* ws = (char*)d_ws;
    size_t off = 0;
    auto alloc = [&](size_t bytes) -> void* {
        off = (off + 255) & ~(size_t)255;
        void* p = ws + off;
        off += bytes;
        return p;
    };
    const size_t M = (size_t)B_ * S_;              // 8192 rows
    _Float16* h16    = (_Float16*)alloc(M * D_ * 2);
    _Float16* xq16   = (_Float16*)alloc(M * KQ_ * 2);
    _Float16* xk16   = (_Float16*)alloc(M * KK_ * 2);
    _Float16* wq16   = (_Float16*)alloc((size_t)D_ * KQ_ * 2);
    _Float16* wk16   = (_Float16*)alloc((size_t)D_ * KK_ * 2);
    _Float16* wv16   = (_Float16*)alloc((size_t)D_ * D_ * 2);
    _Float16* wout16 = (_Float16*)alloc((size_t)D_ * D_ * 2);
    _Float16* scn16  = (_Float16*)alloc(M * NS_ * 2);
    _Float16* ep16   = (_Float16*)alloc(M * NS_ * 2);
    _Float16* q16    = (_Float16*)alloc(M * D_ * 2);
    _Float16* k16    = (_Float16*)alloc(M * D_ * 2);
    _Float16* vt16   = (_Float16*)alloc(M * D_ * 2);  // [B][H][HD][S]
    float*    biasB  = (float*)   alloc((size_t)B_ * S_ * S_ * 4);
    _Float16* o16    = (_Float16*)alloc(M * D_ * 2);
    float*    zbuf   = (float*)   alloc(M * D_ * 4);

    const int T = 256;
    {
        int n = (int)(M * D_);
        cvt_f16_kernel<<<(n + T - 1) / T, T, 0, stream>>>(h, h16, n);
        n = D_ * KQ_;
        cvt_f16_kernel<<<(n + T - 1) / T, T, 0, stream>>>(Wq, wq16, n);
        n = D_ * KK_;
        cvt_f16_kernel<<<(n + T - 1) / T, T, 0, stream>>>(Wk, wk16, n);
        n = D_ * D_;
        cvt_f16_kernel<<<(n + T - 1) / T, T, 0, stream>>>(Wv, wv16, n);
        cvt_f16_kernel<<<(n + T - 1) / T, T, 0, stream>>>(Wout, wout16, n);
        n = (int)(M * KQ_);
        pack_xq_kernel<<<(n + T - 1) / T, T, 0, stream>>>(h, scn, gv, xq16, n);
        n = (int)(M * KK_);
        pack_xk_kernel<<<(n + T - 1) / T, T, 0, stream>>>(h, scn, xk16, n);
        ep_scn_kernel<<<(int)M, 32, 0, stream>>>(scn, gv, scn16, ep16);
    }
    {
        dim3 blk(128);
        dim3 grd(D_ / 64, (int)(M / 64));
        wmma_gemm_xwT<_Float16, 0><<<grd, blk, 0, stream>>>(xq16, wq16, q16, nullptr,
                                                            (int)M, D_, KQ_);
        wmma_gemm_xwT<_Float16, 0><<<grd, blk, 0, stream>>>(xk16, wk16, k16, nullptr,
                                                            (int)M, D_, KK_);
        wmma_gemm_xwT<_Float16, 1><<<grd, blk, 0, stream>>>(h16, wv16, vt16, nullptr,
                                                            (int)M, D_, D_);
    }
    {
        dim3 blk(128);
        dim3 grd(S_ / 64, S_ / 64, B_);
        bias_kernel<<<grd, blk, 0, stream>>>(scn16, ep16, mg, nav, biasB);
    }
    {
        dim3 blk(128);
        dim3 grd(S_ / 64, H_, B_);
        flash_attn_kernel<<<grd, blk, 0, stream>>>(q16, k16, vt16, biasB, mask, o16);
    }
    {
        dim3 blk(128);
        dim3 grd(D_ / 64, (int)(M / 64));
        wmma_gemm_xwT<float, 0><<<grd, blk, 0, stream>>>(o16, wout16, zbuf, h,
                                                         (int)M, D_, D_);
    }
    ln_kernel<<<(int)M, 256, 0, stream>>>(zbuf, ln_g, ln_b, out);
}
